// Head_27238682591852
// MI455X (gfx1250) — matrix operand order repaired and checked by reading, not compiled, not run on hardware
//
#include <hip/hip_runtime.h>
#include <hip/hip_bf16.h>

// ---------------------------------------------------------------------------
// Single-head causal attention, B=8 T=4096 C=1024 HS=128, fp32 in/out.
// Kernel 0: W (f32) -> f16 transposed [n][k] in workspace (once, 768 KB).
// Kernel 1: QKV projection via v_wmma_f32_16x16x32_f16; W chunks async-DMA'd
//           to LDS (double-buffered), output tile LDS-staged -> b128 stores.
// Kernel 2: flash attention via v_wmma_f32_16x16x32_f16; K/V double-buffered
//           with global_load_async_to_lds_b128 (DMA overlaps WMMA compute),
//           LDS-staged coalesced epilogue.
// Fragments are preloaded into distinct registers before each WMMA chain so
// the ds_load_b128s overlap instead of serializing on s_wait_dscnt 0.
// Workspace: qh/kh/vT (24 MiB) + WtG (768 KiB).
// ---------------------------------------------------------------------------

typedef __attribute__((ext_vector_type(16))) _Float16 v16h;
typedef __attribute__((ext_vector_type(8)))  _Float16 v8h;
typedef __attribute__((ext_vector_type(8)))  float    v8f;

#define B_DIM 8
#define T_DIM 4096
#define C_DIM 1024
#define HS    128

static __device__ __forceinline__ v8f wmma_f16(v16h a, v16h b, v8f c) {
  return __builtin_amdgcn_wmma_f32_16x16x32_f16(false, a, false, b, (short)0, c,
                                                false, false);
}

// CDNA5 async DMA: global -> LDS, 16 bytes per lane, tracked by ASYNCcnt.
static __device__ __forceinline__ void async_load_b128(void* lds_dst,
                                                       const void* gsrc) {
  unsigned int lds = (unsigned int)(unsigned long long)lds_dst;
  unsigned long long ga = (unsigned long long)gsrc;
  asm volatile("global_load_async_to_lds_b128 %0, %1, off"
               :
               : "v"(lds), "v"(ga)
               : "memory");
}

static __device__ __forceinline__ void wait_asynccnt0() {
  asm volatile("s_wait_asynccnt 0x0" ::: "memory");
}

// ---------------------------------------------------------------------------
// Kernel 0: convert W (f32 [k][n]) to f16 transposed WtG[which][n][k].
// ---------------------------------------------------------------------------
__global__ __launch_bounds__(256, 1) void w_to_f16t_kernel(
    const float* __restrict__ Wq, const float* __restrict__ Wk,
    const float* __restrict__ Wv, _Float16* __restrict__ WtG) {
  const int which = blockIdx.y;
  const float* __restrict__ W = (which == 0) ? Wq : (which == 1) ? Wk : Wv;
  _Float16* __restrict__ dst = WtG + (size_t)which * HS * C_DIM;
  const int id = blockIdx.x * 256 + threadIdx.x;  // 0 .. 131071
  const int k = id >> 7;   // 0..1023
  const int n = id & 127;  // 0..127
  dst[(size_t)n * C_DIM + k] = (_Float16)W[(size_t)k * HS + n];
}

// ---------------------------------------------------------------------------
// Kernel 1: one projection per blockIdx.y (0=q,1=k,2=v).
// 8 waves; wave owns a 16-row strip x 128 cols (8 accumulator tiles).
// LDS map (halves): [0,4096) WtBuf0, [4096,8192) WtBuf1, [8192,24576) Otile.
// ---------------------------------------------------------------------------
__global__ __launch_bounds__(256, 1) void qkv_proj_kernel(
    const float* __restrict__ x, const _Float16* __restrict__ WtG,
    _Float16* __restrict__ qh, _Float16* __restrict__ kh,
    _Float16* __restrict__ vT) {
  __shared__ __align__(32) char psmem[49152];  // 48 KB
  _Float16* hb = (_Float16*)psmem;

  const int tid  = threadIdx.x;
  const int wave = tid >> 5;
  const int lane = tid & 31;
  const int ln   = lane & 15;
  const int hi   = lane >> 4;
  const int which = blockIdx.y;
  const _Float16* __restrict__ Wt_g = WtG + (size_t)which * HS * C_DIM;

  const int rowBase = blockIdx.x * 128;
  const int gRow    = rowBase + wave * 16 + ln;

  // Stage one 128x32 Wt chunk: 512 b128 chunks, 2 per thread.
  auto stage_w = [&](int k0, int bufi) {
    _Float16* dstb = hb + (bufi ? 4096 : 0);
#pragma unroll
    for (int e = 0; e < 2; ++e) {
      const int c = tid + 256 * e;  // 0..511
      const int n = c >> 2;         // 0..127
      const int part = c & 3;       // chunk of 8 halves
      async_load_b128(&dstb[n * 32 + part * 8],
                      &Wt_g[(size_t)n * C_DIM + k0 + part * 8]);
    }
  };

  v8f acc[8];
#pragma unroll
  for (int i = 0; i < 8; ++i) acc[i] = {};

  stage_w(0, 0);
  wait_asynccnt0();
  __syncthreads();

  for (int k0 = 0; k0 < C_DIM; k0 += 32) {
    const int cur = (k0 >> 5) & 1;
    if (k0 + 32 < C_DIM) stage_w(k0 + 32, cur ^ 1);  // overlaps WMMAs below

    // A fragment: load four independent float4s, then convert f32 -> f16.
    const float* xr = x + (size_t)gRow * C_DIM + k0 + hi * 16;
    const float4 f0 = *(const float4*)(xr + 0);
    const float4 f1 = *(const float4*)(xr + 4);
    const float4 f2 = *(const float4*)(xr + 8);
    const float4 f3 = *(const float4*)(xr + 12);
    v16h a;
    a[0]  = (_Float16)f0.x; a[1]  = (_Float16)f0.y;
    a[2]  = (_Float16)f0.z; a[3]  = (_Float16)f0.w;
    a[4]  = (_Float16)f1.x; a[5]  = (_Float16)f1.y;
    a[6]  = (_Float16)f1.z; a[7]  = (_Float16)f1.w;
    a[8]  = (_Float16)f2.x; a[9]  = (_Float16)f2.y;
    a[10] = (_Float16)f2.z; a[11] = (_Float16)f2.w;
    a[12] = (_Float16)f3.x; a[13] = (_Float16)f3.y;
    a[14] = (_Float16)f3.z; a[15] = (_Float16)f3.w;

    // Preload ALL 8 B-fragments (distinct registers -> overlapping ds_loads),
    // then run the 8-WMMA chain back to back.
    const _Float16* Wl = hb + (cur ? 4096 : 0);
    v16h bf[8];
#pragma unroll
    for (int nt = 0; nt < 8; ++nt)
      bf[nt] = *(const v16h*)&Wl[(nt * 16 + ln) * 32 + hi * 16];
#pragma unroll
    for (int nt = 0; nt < 8; ++nt) acc[nt] = wmma_f16(a, bf[nt], acc[nt]);

    wait_asynccnt0();  // next chunk's DMA (ran under the WMMAs)
    __syncthreads();
  }

  // C-fragment (row r+8*hi, col nt*16+ln) -> LDS output tile.
  _Float16* Otile = hb + 8192;
#pragma unroll
  for (int nt = 0; nt < 8; ++nt)
#pragma unroll
    for (int r = 0; r < 8; ++r)
      Otile[(wave * 16 + r + 8 * hi) * HS + nt * 16 + ln] =
          (_Float16)acc[nt][r];
  __syncthreads();

  // Coalesced global stores: 2048 b128 chunks, 8 per thread.
  if (which != 2) {
    _Float16* __restrict__ dst = (which == 0) ? qh : kh;
#pragma unroll
    for (int e = 0; e < 8; ++e) {
      const int c = tid + 256 * e;
      const int row  = c >> 4;
      const int part = c & 15;
      *(v8h*)&dst[(size_t)(rowBase + row) * HS + part * 8] =
          *(const v8h*)&Otile[row * HS + part * 8];
    }
  } else {
    // vT[b][d][t]: tile column d -> 128 contiguous t positions.
    const int bb = rowBase >> 12;
    const int t0 = rowBase & (T_DIM - 1);
#pragma unroll
    for (int e = 0; e < 8; ++e) {
      const int c = tid + 256 * e;
      const int d  = c >> 4;
      const int tp = c & 15;
      v8h vv;
#pragma unroll
      for (int i = 0; i < 8; ++i) vv[i] = Otile[(tp * 8 + i) * HS + d];
      *(v8h*)&vT[((size_t)bb * HS + d) * T_DIM + t0 + tp * 8] = vv;
    }
  }
}

// ---------------------------------------------------------------------------
// Kernel 2: flash attention. 8 waves per block; one 128-row query tile per
// block, one batch per blockIdx.y. K/V blocks double-buffered via async DMA.
// LDS map (halves): K0 [0,8192) K1 [8192,16384) V0 [16384,24576)
//                   V1 [24576,32768) Pbuf [32768,40960). 80 KB total.
// Epilogue reuses [0,32768) as a 128x128 f32 tile.
// ---------------------------------------------------------------------------
__global__ __launch_bounds__(256, 1) void flash_attn_kernel(
    const _Float16* __restrict__ qh, const _Float16* __restrict__ kh,
    const _Float16* __restrict__ vT, float* __restrict__ out) {
  __shared__ __align__(32) char smem[81920];  // 80 KB
  _Float16* hbase = (_Float16*)smem;

  const int tid  = threadIdx.x;
  const int wave = tid >> 5;
  const int lane = tid & 31;
  const int ln   = lane & 15;
  const int hi   = lane >> 4;
  const int qt   = blockIdx.x;
  const int b    = blockIdx.y;
  const int qBase = qt * 128;
  const int qs    = qBase + wave * 16;
  const float scale = 0.08838834764831845f;  // 1/sqrt(HS)

  // Stage one 64-key K/V block: 2048 b128 chunks total, 8 per thread.
  auto stage_kv = [&](int kstart, int bufi) {
    _Float16* Kd = hbase + (bufi ? 8192 : 0);
    _Float16* Vd = hbase + 16384 + (bufi ? 8192 : 0);
#pragma unroll
    for (int e = 0; e < 4; ++e) {
      const int c   = tid + 256 * e;
      const int key = c >> 4;
      const int dp  = c & 15;
      async_load_b128(&Kd[key * HS + dp * 8],
                      &kh[((size_t)b * T_DIM + kstart + key) * HS + dp * 8]);
    }
#pragma unroll
    for (int e = 0; e < 4; ++e) {
      const int c  = tid + 256 * e;
      const int d  = c >> 3;
      const int kp = c & 7;
      async_load_b128(&Vd[d * 64 + kp * 8],
                      &vT[((size_t)b * HS + d) * T_DIM + kstart + kp * 8]);
    }
  };

  // Preload Q fragments: 16 rows x 128 d as 4 A-fragments (registers).
  v16h qf[4];
  const _Float16* qrow = qh + ((size_t)b * T_DIM + qs + ln) * HS;
#pragma unroll
  for (int ks = 0; ks < 4; ++ks) {
    const int off = ks * 32 + hi * 16;
    const v8h lo  = *(const v8h*)&qrow[off];
    const v8h hi8 = *(const v8h*)&qrow[off + 8];
#pragma unroll
    for (int i = 0; i < 8; ++i) { qf[ks][i] = lo[i]; qf[ks][8 + i] = hi8[i]; }
  }

  v8f Oacc[8];
#pragma unroll
  for (int i = 0; i < 8; ++i) Oacc[i] = {};
  float m_run[8], l_run[8];
#pragma unroll
  for (int r = 0; r < 8; ++r) { m_run[r] = -INFINITY; l_run[r] = 0.0f; }

  const int nblocks = 2 * qt + 2;  // keys 0 .. qBase+127
  stage_kv(0, 0);
  wait_asynccnt0();
  __syncthreads();

  for (int kb = 0; kb < nblocks; ++kb) {
    const int kstart = kb * 64;
    const int cur = kb & 1;
    if (kb + 1 < nblocks) stage_kv((kb + 1) * 64, cur ^ 1);  // overlap

    // S = Q * K^T : per K-step preload 4 B-fragments, then 4 WMMAs.
    v8f S[4];
#pragma unroll
    for (int nt = 0; nt < 4; ++nt) S[nt] = {};
    const _Float16* Kl = hbase + (cur ? 8192 : 0);
#pragma unroll
    for (int ks = 0; ks < 4; ++ks) {
      v16h bf[4];
#pragma unroll
      for (int nt = 0; nt < 4; ++nt)
        bf[nt] = *(const v16h*)&Kl[(nt * 16 + ln) * HS + ks * 32 + hi * 16];
#pragma unroll
      for (int nt = 0; nt < 4; ++nt) S[nt] = wmma_f16(qf[ks], bf[nt], S[nt]);
    }

    // Scale + causal mask + online softmax (row = r+8*hi across lanes 0..15).
#pragma unroll
    for (int r = 0; r < 8; ++r) {
      const int qrow_abs = qs + r + 8 * hi;
      float mx = -INFINITY;
#pragma unroll
      for (int nt = 0; nt < 4; ++nt) {
        const int key = kstart + nt * 16 + ln;
        float v = S[nt][r] * scale;
        v = (key > qrow_abs) ? -INFINITY : v;
        S[nt][r] = v;
        mx = fmaxf(mx, v);
      }
#pragma unroll
      for (int off = 1; off < 16; off <<= 1)
        mx = fmaxf(mx, __shfl_xor(mx, off, 32));
      const float m_new = fmaxf(m_run[r], mx);
      const float alpha = __expf(m_run[r] - m_new);
      m_run[r] = m_new;
      float rs = 0.0f;
#pragma unroll
      for (int nt = 0; nt < 4; ++nt) {
        const float p = __expf(S[nt][r] - m_new);
        S[nt][r] = p;
        rs += p;
      }
#pragma unroll
      for (int off = 1; off < 16; off <<= 1) rs += __shfl_xor(rs, off, 32);
      l_run[r] = l_run[r] * alpha + rs;
#pragma unroll
      for (int dt = 0; dt < 8; ++dt) Oacc[dt][r] *= alpha;
    }

    // P (C-layout f32) -> f16 LDS -> A-fragments (per-wave private region).
    _Float16* pb = hbase + 32768 + wave * (16 * 64);
#pragma unroll
    for (int nt = 0; nt < 4; ++nt)
#pragma unroll
      for (int r = 0; r < 8; ++r)
        pb[(r + 8 * hi) * 64 + nt * 16 + ln] = (_Float16)S[nt][r];
    __builtin_amdgcn_wave_barrier();
    asm volatile("s_wait_dscnt 0" ::: "memory");

    v16h pf[2];
#pragma unroll
    for (int ks = 0; ks < 2; ++ks) {
      const int off = ks * 32 + hi * 16;
      const v8h lo  = *(const v8h*)&pb[ln * 64 + off];
      const v8h hi8 = *(const v8h*)&pb[ln * 64 + off + 8];
#pragma unroll
      for (int i = 0; i < 8; ++i) { pf[ks][i] = lo[i]; pf[ks][8 + i] = hi8[i]; }
    }

    // O += P * V : per K-step, two groups of 4 preloaded V-fragments.
    const _Float16* Vl = hbase + 16384 + (cur ? 8192 : 0);
#pragma unroll
    for (int ks = 0; ks < 2; ++ks) {
#pragma unroll
      for (int g = 0; g < 2; ++g) {
        v16h vf[4];
#pragma unroll
        for (int j = 0; j < 4; ++j) {
          const int dt = g * 4 + j;
          vf[j] = *(const v16h*)&Vl[(dt * 16 + ln) * 64 + ks * 32 + hi * 16];
        }
#pragma unroll
        for (int j = 0; j < 4; ++j) {
          const int dt = g * 4 + j;
          Oacc[dt] = wmma_f16(pf[ks], vf[j], Oacc[dt]);
        }
      }
    }

    wait_asynccnt0();  // next block's DMA (ran under the WMMAs)
    __syncthreads();
  }

  // Epilogue: normalize (one reciprocal per row), stage 128x128 f32 tile in
  // LDS (reusing K/V buffers), then coalesced b128 stores (16 per thread).
  float* ot = (float*)smem;
#pragma unroll
  for (int r = 0; r < 8; ++r) {
    const float invl = 1.0f / l_run[r];
#pragma unroll
    for (int dt = 0; dt < 8; ++dt)
      ot[(wave * 16 + r + 8 * hi) * HS + dt * 16 + ln] = Oacc[dt][r] * invl;
  }
  __syncthreads();
#pragma unroll
  for (int e = 0; e < 16; ++e) {
    const int c = tid + 256 * e;  // 0..4095 float4 chunks
    const int row  = c >> 5;
    const int part = c & 31;
    *(float4*)&out[((size_t)b * T_DIM + qBase + row) * HS + part * 4] =
        *(const float4*)&ot[row * HS + part * 4];
  }
}

// ---------------------------------------------------------------------------
extern "C" void kernel_launch(void* const* d_in, const int* in_sizes, int n_in,
                              void* d_out, int out_size, void* d_ws,
                              size_t ws_size, hipStream_t stream) {
  const float* x  = (const float*)d_in[0];
  const float* Wq = (const float*)d_in[1];
  const float* Wk = (const float*)d_in[2];
  const float* Wv = (const float*)d_in[3];
  float* out = (float*)d_out;

  const size_t elems = (size_t)B_DIM * T_DIM * HS;  // 4,194,304
  _Float16* qh  = (_Float16*)d_ws;                  // 8 MiB
  _Float16* kh  = qh + elems;                       // 8 MiB
  _Float16* vT  = kh + elems;                       // 8 MiB (transposed V)
  _Float16* WtG = vT + elems;                       // 768 KiB (f16 W^T x3)
  (void)ws_size;  // requires >= ~24.8 MiB

  dim3 wgrid((C_DIM * HS) / 256, 3);
  w_to_f16t_kernel<<<wgrid, 256, 0, stream>>>(Wq, Wk, Wv, WtG);

  dim3 pgrid((B_DIM * T_DIM) / 128, 3);  // 256 row-tiles x {q,k,v}
  qkv_proj_kernel<<<pgrid, 256, 0, stream>>>(x, WtG, qh, kh, vT);

  dim3 agrid(T_DIM / 128, B_DIM);  // 32 query tiles x 8 batches
  flash_attn_kernel<<<agrid, 256, 0, stream>>>(qh, kh, vT, out);
}
